// Network_71674414236217
// MI455X (gfx1250) — matrix operand order repaired and checked by reading, not compiled, not run on hardware
//
#include <hip/hip_runtime.h>
#include <stdint.h>

// ---------------------------------------------------------------------------
// CDNA5 WMMA (wave32): D(16x16 f32) = A(16x32 bf16) x B(32x16 bf16) + C
// ---------------------------------------------------------------------------
typedef __attribute__((ext_vector_type(16))) __bf16   v16bf;
typedef __attribute__((ext_vector_type(8)))  float    v8f;
typedef __attribute__((ext_vector_type(8)))  uint32_t v8u;

union FragU { v8u u; v16bf b; };

__device__ __forceinline__ v8f zero8() {
  v8f z;
#pragma unroll
  for (int i = 0; i < 8; ++i) z[i] = 0.0f;
  return z;
}

__device__ __forceinline__ uint32_t bf16_1(float f) {
  uint32_t x = __float_as_uint(f);
  return (x + 0x7FFFu + ((x >> 16) & 1u)) >> 16;   // RNE f32->bf16
}
__device__ __forceinline__ uint32_t pack2(float a, float b) {
  return bf16_1(a) | (bf16_1(b) << 16);
}

__device__ __forceinline__ v8f wmma_bf16(v16bf a, v16bf b, v8f c) {
  return __builtin_amdgcn_wmma_f32_16x16x32_bf16(
      false, a, false, b, (short)0, c, false, false);
}

// Load a 16x32 bf16 A-fragment stored pre-swizzled in LDS:
// dword index = (kt*32 + lane)*8 + v  ->  two ds_load_b128 per lane.
__device__ __forceinline__ v16bf load_a_bf(const uint32_t* base, int kt, int lane) {
  const uint4* q = (const uint4*)(base + (size_t)(kt * 32 + lane) * 8);
  uint4 lo = q[0], hi = q[1];
  FragU c;
  c.u[0] = lo.x; c.u[1] = lo.y; c.u[2] = lo.z; c.u[3] = lo.w;
  c.u[4] = hi.x; c.u[5] = hi.y; c.u[6] = hi.z; c.u[7] = hi.w;
  return c.b;
}

// B-matrix fragment from prepacked global arena (hoisted to VGPRs by compiler).
__device__ __forceinline__ v16bf load_b_frag(const uint32_t* wf, int frag, int lane) {
  const uint4* q = (const uint4*)(wf + ((size_t)frag * 32 + (size_t)lane) * 8);
  uint4 lo = q[0], hi = q[1];
  FragU c;
  c.u[0] = lo.x; c.u[1] = lo.y; c.u[2] = lo.z; c.u[3] = lo.w;
  c.u[4] = hi.x; c.u[5] = hi.y; c.u[6] = hi.z; c.u[7] = hi.w;
  return c.b;
}

// Halfword index (m=0) inside a bf16 A-layout state tile for column `col`.
// Element (m, col) lives at hw = ((kt*32 + m + 16h)*8 + v)*2 + (col&1).
__device__ __forceinline__ int state_hw_index(int col) {
  int kt = col >> 5;
  int kl = col & 31;
  int h  = (kl >> 3) & 1;           // lane-half that owns this k
  int kp = kl - 8 * h;              // {0..7} or {16..23}
  int v  = (kp < 8) ? (kp >> 1) : (((kp - 16) >> 1) + 4);
  return ((kt * 32 + 16 * h) * 8 + v) * 2 + (kl & 1);
}

// Write 8 C-fragment elements (rows m = r + 8*halfw, fixed col) as bf16
// into an A-layout LDS tile. Per element: ~3 VALU + ds_store_b16 (imm offset).
__device__ __forceinline__ void store8_bf16(uint32_t* base, int hwBase, int halfw,
                                            const float* vals) {
  uint16_t* bp = (uint16_t*)base + hwBase + 8 * halfw * 16;
#pragma unroll
  for (int r = 0; r < 8; ++r) {
    uint32_t x = __float_as_uint(vals[r]);
    uint32_t rr = x + 0x7FFFu + ((x >> 16) & 1u);
    bp[r * 16] = (uint16_t)(rr >> 16);
  }
}

// Convert one 16x32 K-tile of this lane's x row into A-fragment dwords in LDS.
__device__ __forceinline__ void conv_x_tile(uint32_t* dstBase, const float* xt,
                                            int kt, int lane, int half) {
  uint32_t* dst = dstBase + (size_t)(kt * 32 + lane) * 8;
  const float* pk = xt + kt * 32 + 8 * half;
#pragma unroll
  for (int v = 0; v < 8; ++v) {
    int k = (v < 4) ? (2 * v) : (2 * v + 8);
    float2 f = *(const float2*)(pk + k);
    dst[v] = pack2(f.x, f.y);
  }
}

__device__ __forceinline__ float sigm(float x) {
  return __builtin_amdgcn_rcpf(1.0f + __expf(-x));
}
__device__ __forceinline__ float tanh_fast(float x) {
  // tanh(x) = 1 - 2/(1+exp(2x)); exp->inf gives 1, exp->0 gives -1.
  float e = __expf(2.0f * x);
  return 1.0f - 2.0f * __builtin_amdgcn_rcpf(1.0f + e);
}

// ---------------------------------------------------------------------------
// Weight fragment arena in d_ws (dword offsets)
// ---------------------------------------------------------------------------
#define OFF_VWI 0       // van_wi  256x32 : KT=8,  NT=2
#define OFF_VWH 4096    // van_wh   32x32 : KT=1,  NT=2
#define OFF_L1  4608    // lstm1_w 96x256 : KT=3,  NT=16
#define OFF_L2  16896   // lstm2_w 128x256: KT=4,  NT=16
#define OFF_GI  33280   // gru_wi  64x192 : KT=2,  NT=12
#define OFF_GH  39424   // gru_wh  64x192 : KT=2,  NT=12

__global__ void pack_weights(const float* __restrict__ src, uint32_t* __restrict__ dst,
                             int K, int N, int KT, int NT) {
  int idx = blockIdx.x * blockDim.x + threadIdx.x;
  int total = KT * NT * 256;
  if (idx >= total) return;
  int v    = idx & 7;
  int lane = (idx >> 3) & 31;
  int frag = idx >> 8;
  int kt = frag % KT;
  int nt = frag / KT;
  int half = lane >> 4;
  int n  = nt * 16 + (lane & 15);
  int k0 = kt * 32 + 8 * half + 2 * v + (v >= 4 ? 8 : 0);
  float f0 = (k0     < K && n < N) ? src[(size_t)k0       * N + n] : 0.0f;
  float f1 = (k0 + 1 < K && n < N) ? src[(size_t)(k0 + 1) * N + n] : 0.0f;
  dst[idx] = pack2(f0, f1);
}

// ---------------------------------------------------------------------------
// Main recurrent kernel: 32 blocks x 128 threads (4 waves). Block b owns batch
// rows [16b,16b+16); wave w owns gate-column tiles {w, w+4, w+8, w+12}.
// Recurrent state lives in LDS as bf16 in A-fragment layout, double-buffered.
// c1/c2/h3 (wave-owned columns) stay in registers as f32.
// ---------------------------------------------------------------------------
__global__ __launch_bounds__(128, 1) void rnn_main(
    const float* __restrict__ x,
    const float* __restrict__ van_bi, const float* __restrict__ van_bh,
    const float* __restrict__ lstm1_b, const float* __restrict__ lstm2_b,
    const float* __restrict__ gru_b,
    const uint32_t* __restrict__ wf,
    const float* __restrict__ mlp_w1, const float* __restrict__ mlp_b1,
    const float* __restrict__ mlp_w2, const float* __restrict__ mlp_b2,
    const float* __restrict__ mlp_w3, const float* __restrict__ mlp_b3,
    const float* __restrict__ sf,
    float* __restrict__ out) {
  constexpr int T = 512, D = 256;

  __shared__ __align__(16) uint32_t s_x [2][8 * 32 * 8];   // x_t bf16 A-frags
  __shared__ __align__(16) uint32_t s_h0[2][1 * 32 * 8];
  __shared__ __align__(16) uint32_t s_h1[2][2 * 32 * 8];
  __shared__ __align__(16) uint32_t s_h2[2][2 * 32 * 8];
  __shared__ __align__(16) uint32_t s_h3[2][2 * 32 * 8];
  __shared__ __align__(16) uint32_t s_rh3[2 * 32 * 8];
  __shared__ __align__(16) float    s_hf[16 * 64];         // final h3 f32 (tail)

  const int tid  = threadIdx.x;
  const int wv   = tid >> 5;      // wave 0..3 (uniform in wave)
  const int lane = tid & 31;
  const int n    = lane & 15;
  const int half = lane >> 4;
  const int col  = 16 * wv + n;   // owned column in 64-wide state

  const int hwS  = state_hw_index(col);            // h1/h2/h3/rh3 tiles
  const int hwH0 = state_hw_index(16 * wv + n);    // h0 tile (waves 0,1)

  // zero-init the "read" buffers for t=0
  for (int i = tid; i < 256; i += 128) s_h0[0][i] = 0u;
  for (int i = tid; i < 512; i += 128) { s_h1[0][i] = 0u; s_h2[0][i] = 0u; s_h3[0][i] = 0u; }

  // loop-invariant per-lane biases
  float b0 = 0.0f;
  if (wv < 2) b0 = van_bi[16 * wv + n] + van_bh[16 * wv + n];
  const float b1i = lstm1_b[      col], b1g = lstm1_b[ 64 + col],
              b1f = lstm1_b[128 + col], b1o = lstm1_b[192 + col];
  const float b2i = lstm2_b[      col], b2g = lstm2_b[ 64 + col],
              b2f = lstm2_b[128 + col], b2o = lstm2_b[192 + col];
  const float bz = gru_b[col], br = gru_b[64 + col], ba = gru_b[128 + col];

  // registers: per-wave-owned state columns
  float c1r[8], c2r[8], h3r[8];
#pragma unroll
  for (int r = 0; r < 8; ++r) { c1r[r] = 0.0f; c2r[r] = 0.0f; h3r[r] = 0.0f; }

  // per-lane x row pointer (row m = lane&15 of this block's batch tile)
  const float* xrow = x + (size_t)(blockIdx.x * 16 + n) * (size_t)(T * D);

  // pre-convert x(0) into buffer 0 (wave w does K-tiles {w, w+4})
  conv_x_tile(s_x[0], xrow, wv,     lane, half);
  conv_x_tile(s_x[0], xrow, wv + 4, lane, half);
  __syncthreads();

  for (int t = 0; t < T; ++t) {
    const int p = t & 1;

    // ---- Phase A: h0 = relu(x_t @ van_wi + h0 @ van_wh + b)  (waves 0,1) ---
    float h0v[8];
    if (wv < 2) {
      v8f acc0 = zero8(), acc1 = zero8();
#pragma unroll
      for (int kt = 0; kt < 8; kt += 2) {
        acc0 = wmma_bf16(load_a_bf(s_x[p], kt,     lane),
                         load_b_frag(wf + OFF_VWI, wv * 8 + kt,     lane), acc0);
        acc1 = wmma_bf16(load_a_bf(s_x[p], kt + 1, lane),
                         load_b_frag(wf + OFF_VWI, wv * 8 + kt + 1, lane), acc1);
      }
      acc0 = wmma_bf16(load_a_bf(s_h0[p], 0, lane),
                       load_b_frag(wf + OFF_VWH, wv, lane), acc0);
#pragma unroll
      for (int r = 0; r < 8; ++r) h0v[r] = fmaxf(acc0[r] + acc1[r] + b0, 0.0f);
      store8_bf16(s_h0[1 - p], hwH0, half, h0v);   // different buffer: no pre-barrier
    }
    __syncthreads();

    // ---- overlap: convert x(t+1) into s_x[1-p] (all 4 waves) ----
    if (t + 1 < T) {
      const float* xnext = xrow + (size_t)(t + 1) * D;
      conv_x_tile(s_x[1 - p], xnext, wv,     lane, half);
      conv_x_tile(s_x[1 - p], xnext, wv + 4, lane, half);
      if (t + 2 < T) __builtin_prefetch(xrow + (size_t)(t + 2) * D, 0, 1);
    }

    // ---- Phase B: LSTM1, [h0,h1](16x96) @ W(96x256) ----
    {
      v16bf aK0 = load_a_bf(s_h0[1 - p], 0, lane);
      v16bf aK1 = load_a_bf(s_h1[p],     0, lane);
      v16bf aK2 = load_a_bf(s_h1[p],     1, lane);
      const uint32_t* w1 = wf + OFF_L1;
      v8f gi = zero8(), gg = zero8(), gf = zero8(), go = zero8();
      gi = wmma_bf16(aK0, load_b_frag(w1, (wv +  0) * 3 + 0, lane), gi);
      gi = wmma_bf16(aK1, load_b_frag(w1, (wv +  0) * 3 + 1, lane), gi);
      gi = wmma_bf16(aK2, load_b_frag(w1, (wv +  0) * 3 + 2, lane), gi);
      gg = wmma_bf16(aK0, load_b_frag(w1, (wv +  4) * 3 + 0, lane), gg);
      gg = wmma_bf16(aK1, load_b_frag(w1, (wv +  4) * 3 + 1, lane), gg);
      gg = wmma_bf16(aK2, load_b_frag(w1, (wv +  4) * 3 + 2, lane), gg);
      gf = wmma_bf16(aK0, load_b_frag(w1, (wv +  8) * 3 + 0, lane), gf);
      gf = wmma_bf16(aK1, load_b_frag(w1, (wv +  8) * 3 + 1, lane), gf);
      gf = wmma_bf16(aK2, load_b_frag(w1, (wv +  8) * 3 + 2, lane), gf);
      go = wmma_bf16(aK0, load_b_frag(w1, (wv + 12) * 3 + 0, lane), go);
      go = wmma_bf16(aK1, load_b_frag(w1, (wv + 12) * 3 + 1, lane), go);
      go = wmma_bf16(aK2, load_b_frag(w1, (wv + 12) * 3 + 2, lane), go);
      float h1v[8];
#pragma unroll
      for (int r = 0; r < 8; ++r) {
        float iv = sigm(gi[r] + b1i);
        float gv = tanh_fast(gg[r] + b1g);
        float fv = sigm(gf[r] + b1f + 1.0f);
        float ov = sigm(go[r] + b1o);
        float c  = fv * c1r[r] + iv * gv;
        c1r[r]   = c;
        h1v[r]   = ov * tanh_fast(c);
      }
      store8_bf16(s_h1[1 - p], hwS, half, h1v);
    }
    __syncthreads();

    // ---- Phase C: LSTM2, [h1,h2](16x128) @ W(128x256) ----
    {
      v16bf a0 = load_a_bf(s_h1[1 - p], 0, lane);
      v16bf a1 = load_a_bf(s_h1[1 - p], 1, lane);
      v16bf a2 = load_a_bf(s_h2[p],     0, lane);
      v16bf a3 = load_a_bf(s_h2[p],     1, lane);
      const uint32_t* w2 = wf + OFF_L2;
      v8f gi = zero8(), gg = zero8(), gf = zero8(), go = zero8();
      gi = wmma_bf16(a0, load_b_frag(w2, (wv +  0) * 4 + 0, lane), gi);
      gi = wmma_bf16(a1, load_b_frag(w2, (wv +  0) * 4 + 1, lane), gi);
      gi = wmma_bf16(a2, load_b_frag(w2, (wv +  0) * 4 + 2, lane), gi);
      gi = wmma_bf16(a3, load_b_frag(w2, (wv +  0) * 4 + 3, lane), gi);
      gg = wmma_bf16(a0, load_b_frag(w2, (wv +  4) * 4 + 0, lane), gg);
      gg = wmma_bf16(a1, load_b_frag(w2, (wv +  4) * 4 + 1, lane), gg);
      gg = wmma_bf16(a2, load_b_frag(w2, (wv +  4) * 4 + 2, lane), gg);
      gg = wmma_bf16(a3, load_b_frag(w2, (wv +  4) * 4 + 3, lane), gg);
      gf = wmma_bf16(a0, load_b_frag(w2, (wv +  8) * 4 + 0, lane), gf);
      gf = wmma_bf16(a1, load_b_frag(w2, (wv +  8) * 4 + 1, lane), gf);
      gf = wmma_bf16(a2, load_b_frag(w2, (wv +  8) * 4 + 2, lane), gf);
      gf = wmma_bf16(a3, load_b_frag(w2, (wv +  8) * 4 + 3, lane), gf);
      go = wmma_bf16(a0, load_b_frag(w2, (wv + 12) * 4 + 0, lane), go);
      go = wmma_bf16(a1, load_b_frag(w2, (wv + 12) * 4 + 1, lane), go);
      go = wmma_bf16(a2, load_b_frag(w2, (wv + 12) * 4 + 2, lane), go);
      go = wmma_bf16(a3, load_b_frag(w2, (wv + 12) * 4 + 3, lane), go);
      float h2v[8];
#pragma unroll
      for (int r = 0; r < 8; ++r) {
        float iv = sigm(gi[r] + b2i);
        float gv = tanh_fast(gg[r] + b2g);
        float fv = sigm(gf[r] + b2f + 1.0f);
        float ov = sigm(go[r] + b2o);
        float c  = fv * c2r[r] + iv * gv;
        c2r[r]   = c;
        h2v[r]   = ov * tanh_fast(c);
      }
      store8_bf16(s_h2[1 - p], hwS, half, h2v);
    }
    __syncthreads();

    // ---- Phase D: GRU ----
    {
      v16bf a2_0 = load_a_bf(s_h2[1 - p], 0, lane);
      v16bf a2_1 = load_a_bf(s_h2[1 - p], 1, lane);
      v16bf a3_0 = load_a_bf(s_h3[p],     0, lane);
      v16bf a3_1 = load_a_bf(s_h3[p],     1, lane);
      const uint32_t* wgi = wf + OFF_GI;
      const uint32_t* wgh = wf + OFF_GH;

      // D1: zr = h2 @ Wi[:, :128] + h3 @ Wh[:, :128] + b
      v8f az = zero8(), arr = zero8();
      az  = wmma_bf16(a2_0, load_b_frag(wgi, (wv + 0) * 2 + 0, lane), az);
      az  = wmma_bf16(a2_1, load_b_frag(wgi, (wv + 0) * 2 + 1, lane), az);
      az  = wmma_bf16(a3_0, load_b_frag(wgh, (wv + 0) * 2 + 0, lane), az);
      az  = wmma_bf16(a3_1, load_b_frag(wgh, (wv + 0) * 2 + 1, lane), az);
      arr = wmma_bf16(a2_0, load_b_frag(wgi, (wv + 4) * 2 + 0, lane), arr);
      arr = wmma_bf16(a2_1, load_b_frag(wgi, (wv + 4) * 2 + 1, lane), arr);
      arr = wmma_bf16(a3_0, load_b_frag(wgh, (wv + 4) * 2 + 0, lane), arr);
      arr = wmma_bf16(a3_1, load_b_frag(wgh, (wv + 4) * 2 + 1, lane), arr);
      float zv[8], rh[8];
#pragma unroll
      for (int r = 0; r < 8; ++r) {
        zv[r]    = sigm(az[r] + bz);
        float rv = sigm(arr[r] + br);
        rh[r]    = rv * h3r[r];
      }
      store8_bf16(s_rh3, hwS, half, rh);   // prior readers fenced by earlier barriers
      __syncthreads();

      // D2: a = tanh(h2 @ Wi[:, 128:] + (r*h3) @ Wh[:, 128:] + b)
      v16bf aR0 = load_a_bf(s_rh3, 0, lane);
      v16bf aR1 = load_a_bf(s_rh3, 1, lane);
      v8f aa = zero8();
      aa = wmma_bf16(a2_0, load_b_frag(wgi, (wv + 8) * 2 + 0, lane), aa);
      aa = wmma_bf16(a2_1, load_b_frag(wgi, (wv + 8) * 2 + 1, lane), aa);
      aa = wmma_bf16(aR0,  load_b_frag(wgh, (wv + 8) * 2 + 0, lane), aa);
      aa = wmma_bf16(aR1,  load_b_frag(wgh, (wv + 8) * 2 + 1, lane), aa);
      float h3v[8];
#pragma unroll
      for (int r = 0; r < 8; ++r) {
        float av = tanh_fast(aa[r] + ba);
        float hn = (1.0f - zv[r]) * h3r[r] + zv[r] * av;
        h3r[r] = hn;
        h3v[r] = hn;
      }
      store8_bf16(s_h3[1 - p], hwS, half, h3v);
      __syncthreads();
    }
  } // t loop

  // ---- Tail: dump final h3 (f32 regs) to LDS, then MLP head + policy ----
#pragma unroll
  for (int r = 0; r < 8; ++r) s_hf[(r + 8 * half) * 64 + col] = h3r[r];
  __syncthreads();

  if (tid < 16) {
    int m = tid;
    float e1[32];
    for (int j = 0; j < 32; ++j) {
      float s = mlp_b1[j];
      for (int k = 0; k < 64; ++k) s += s_hf[m * 64 + k] * mlp_w1[k * 32 + j];
      e1[j] = fmaxf(s, 0.0f);
    }
    float e2[32];
    for (int j = 0; j < 32; ++j) {
      float s = mlp_b2[j];
      for (int k = 0; k < 32; ++k) s += e1[k] * mlp_w2[k * 32 + j];
      e2[j] = fmaxf(s, 0.0f);
    }
    float pol = 0.0f;
    for (int j = 0; j < 12; ++j) {
      float s = mlp_b3[j];
      for (int k = 0; k < 32; ++k) s += e2[k] * mlp_w3[k * 12 + j];
      pol += s * sf[j];   // preference_vectors == 1
    }
    out[24 + blockIdx.x * 16 + m] = pol;
  }
  if (blockIdx.x == 0 && tid >= 32 && tid < 44) {
    int j = tid - 32;
    out[j]      = sf[j];   // successor_features
    out[12 + j] = 1.0f;    // preference_vectors
  }
}

// ---------------------------------------------------------------------------
extern "C" void kernel_launch(void* const* d_in, const int* in_sizes, int n_in,
                              void* d_out, int out_size, void* d_ws, size_t ws_size,
                              hipStream_t stream) {
  (void)in_sizes; (void)n_in; (void)out_size; (void)ws_size;
  const float* x       = (const float*)d_in[0];
  const float* van_wi  = (const float*)d_in[1];
  const float* van_bi  = (const float*)d_in[2];
  const float* van_wh  = (const float*)d_in[3];
  const float* van_bh  = (const float*)d_in[4];
  const float* lstm1_w = (const float*)d_in[5];
  const float* lstm1_b = (const float*)d_in[6];
  const float* lstm2_w = (const float*)d_in[7];
  const float* lstm2_b = (const float*)d_in[8];
  const float* gru_wi  = (const float*)d_in[9];
  const float* gru_wh  = (const float*)d_in[10];
  const float* gru_b   = (const float*)d_in[11];
  const float* mlp_w1  = (const float*)d_in[12];
  const float* mlp_b1  = (const float*)d_in[13];
  const float* mlp_w2  = (const float*)d_in[14];
  const float* mlp_b2  = (const float*)d_in[15];
  const float* mlp_w3  = (const float*)d_in[16];
  const float* mlp_b3  = (const float*)d_in[17];
  const float* sf      = (const float*)d_in[18];

  uint32_t* wfp = (uint32_t*)d_ws;

  auto pk = [&](const float* src, int off, int K, int N, int KT, int NT) {
    int total = KT * NT * 256;
    pack_weights<<<(total + 255) / 256, 256, 0, stream>>>(src, wfp + off, K, N, KT, NT);
  };
  pk(van_wi,  OFF_VWI, 256,  32, 8,  2);
  pk(van_wh,  OFF_VWH,  32,  32, 1,  2);
  pk(lstm1_w, OFF_L1,   96, 256, 3, 16);
  pk(lstm2_w, OFF_L2,  128, 256, 4, 16);
  pk(gru_wi,  OFF_GI,   64, 192, 2, 12);
  pk(gru_wh,  OFF_GH,   64, 192, 2, 12);

  rnn_main<<<dim3(32), dim3(128), 0, stream>>>(
      x, van_bi, van_bh, lstm1_b, lstm2_b, gru_b, wfp,
      mlp_w1, mlp_b1, mlp_w2, mlp_b2, mlp_w3, mlp_b3, sf, (float*)d_out);
}
